// DCCF_12841952215148
// MI455X (gfx1250) — matrix operand order repaired and checked by reading, not compiled, not run on hardware
//
#include <hip/hip_runtime.h>
#include <hip/hip_bf16.h>

// DCCF-style graph model on MI455X (gfx1250, wave32).
//  - Node feature arrays (150000x64 f32 = 38.4MB) fit in 192MB L2 -> random
//    per-edge gathers / atomic scatters are L2-resident; SpMM is thread-per-
//    (edge,feature) with global_atomic_add_f32.
//  - Dense intent block softmax(x@W)@W^T on v_wmma_f32_16x16x32_bf16.
//    Intent matrices are pre-swizzled ONCE into WMMA B-fragment layout in
//    workspace (64KB, L2-resident), so every B fragment is one contiguous
//    32B read per lane (global_load_b128 pair). Softmax probabilities are
//    written to LDS directly in A-fragment layout, so GEMM2 A fragments are
//    contiguous ds_load_b128 pairs, hoisted out of the column-tile loop.

#define DIM 64
#define KK  128
#define NUSERS 90000

typedef __bf16 bf;
typedef __attribute__((ext_vector_type(16))) __bf16 v16bf;
typedef __attribute__((ext_vector_type(8)))  float  v8f;

// ---------------------------------------------------------------- utilities
__global__ void k_zero(float* __restrict__ p, long n) {
  long i = (long)blockIdx.x * blockDim.x + threadIdx.x;
  if (i < n) p[i] = 0.f;
}

__global__ void k_init(const float* __restrict__ emb, float* __restrict__ acc,
                       float* __restrict__ x, long n) {
  long i = (long)blockIdx.x * blockDim.x + threadIdx.x;
  if (i < n) { float v = emb[i]; acc[i] = v; x[i] = v; }
}

// deg[h[e]] += 1
__global__ void k_deg(const int* __restrict__ h, float* __restrict__ deg, int e) {
  int i = blockIdx.x * blockDim.x + threadIdx.x;
  if (i < e) atomicAdd(&deg[h[i]], 1.f);
}

// invn = deg > 0 ? deg^-0.5 : 0
__global__ void k_dinvsqrt(const float* __restrict__ deg, float* __restrict__ invn, int n) {
  int i = blockIdx.x * blockDim.x + threadIdx.x;
  if (i < n) { float d = deg[i]; invn[i] = (d > 0.f) ? rsqrtf(d) : 0.f; }
}

// g_vals[e] = invn[h]*invn[t]
__global__ void k_gvals(const float* __restrict__ invn, const int* __restrict__ h,
                        const int* __restrict__ t, float* __restrict__ gv, int e) {
  int i = blockIdx.x * blockDim.x + threadIdx.x;
  if (i < e) gv[i] = invn[h[i]] * invn[t[i]];
}

// out[h[e]] += vals[e] * x[t[e]]   (thread per (edge, d))
__global__ void k_spmm(const float* __restrict__ vals, const int* __restrict__ h,
                       const int* __restrict__ t, const float* __restrict__ x,
                       float* __restrict__ out, int e) {
  long tid = (long)blockIdx.x * blockDim.x + threadIdx.x;
  if (tid >= (long)e * DIM) return;
  int edge = (int)(tid >> 6);
  int d    = (int)(tid & 63);
  float v  = vals[edge];
  int hh = h[edge], tt = t[edge];
  atomicAdd(&out[(size_t)hh * DIM + d], v * x[(size_t)tt * DIM + d]);
}

// invn[n] = 1 / max(||f[n]||, 1e-12)  (one wave per node, float2 per lane)
__global__ void k_invnorm(const float* __restrict__ f, float* __restrict__ invn, int n) {
  int idx  = blockIdx.x * (blockDim.x >> 5) + (threadIdx.x >> 5);
  int lane = threadIdx.x & 31;
  if (idx >= n) return;
  const float2* r = (const float2*)(f + (size_t)idx * DIM);
  float2 v = r[lane];
  float ss = v.x * v.x + v.y * v.y;
  #pragma unroll
  for (int off = 1; off < 32; off <<= 1) ss += __shfl_xor(ss, off, 32);
  if (lane == 0) {
    float nrm = fmaxf(sqrtf(ss), 1e-12f);
    invn[idx] = 1.f / nrm;
  }
}

// alpha[e] = (cos(f[h],f[t]) + 1)/2 ; rowsum[h] += alpha  (one wave per edge)
__global__ void k_alpha(const float* __restrict__ f, const float* __restrict__ invn,
                        const int* __restrict__ h, const int* __restrict__ t,
                        float* __restrict__ alpha, float* __restrict__ rowsum, int e) {
  int idx  = blockIdx.x * (blockDim.x >> 5) + (threadIdx.x >> 5);
  int lane = threadIdx.x & 31;
  if (idx >= e) return;
  int hh = h[idx], tt = t[idx];
  const float2* fh = (const float2*)(f + (size_t)hh * DIM);
  const float2* ft = (const float2*)(f + (size_t)tt * DIM);
  float2 a = fh[lane], b = ft[lane];
  float d = a.x * b.x + a.y * b.y;
  #pragma unroll
  for (int off = 1; off < 32; off <<= 1) d += __shfl_xor(d, off, 32);
  if (lane == 0) {
    float al = (d * invn[hh] * invn[tt] + 1.f) * 0.5f;
    alpha[idx] = al;
    atomicAdd(&rowsum[hh], al);
  }
}

// alpha[e] *= (rowsum[h] > 0 ? 1/rowsum[h] : 0)
__global__ void k_scale(float* __restrict__ alpha, const float* __restrict__ rowsum,
                        const int* __restrict__ h, int e) {
  int i = blockIdx.x * blockDim.x + threadIdx.x;
  if (i >= e) return;
  float r = rowsum[h[i]];
  alpha[i] = (r > 0.f) ? alpha[i] / r : 0.f;
}

// x_new = (gaa+iaa accumulated in xn) + gnn + intl + x ;  acc += x_new
__global__ void k_combine(float* __restrict__ xn, const float* __restrict__ gnn,
                          const float* __restrict__ intl, const float* __restrict__ x,
                          float* __restrict__ acc, long n) {
  long i = (long)blockIdx.x * blockDim.x + threadIdx.x;
  if (i >= n) return;
  float v = xn[i] + gnn[i] + intl[i] + x[i];
  xn[i] = v;
  acc[i] += v;
}

// ---------------------------------------- pre-swizzle W into B-frag layout
// B1: GEMM1 B frags (W as 64x128), tiles (ks<2, ct<8): lane holds 16
//     contiguous bf16.  B2: GEMM2 B frags (W^T as 128x64), tiles (ks<4, dt<4).
// Per matrix per set: 16 tiles * 32 lanes * 16 halfs = 8192 bf16 = 16KB.
// 16-bit B-frag layout: lanes 0-15 col=lane, elem i<8 -> K=kbase+i, i>=8 -> K=kbase+8+i
// with kbase = (lane>>4)*8 within the 32-deep K chunk.
__global__ void k_prepW(const float* __restrict__ uW, const float* __restrict__ iW,
                        bf* __restrict__ B1, bf* __restrict__ B2) {
  int f = blockIdx.x * blockDim.x + threadIdx.x;
  if (f >= 4 * 8192) return;
  int set  = f >> 14;          // 0: B1, 1: B2
  int rem  = f & 16383;
  int m    = rem >> 13;        // 0: user, 1: item
  int idx  = rem & 8191;       // position within [16 tiles][32 lanes][16]
  int tile = idx >> 9;
  int l    = (idx >> 4) & 31;
  int i    = idx & 15;
  int col  = l & 15;
  int kbase = (l >> 4) * 8 + ((i >= 8) ? 8 : 0);
  const float* W = m ? iW : uW;
  float v;
  if (set == 0) {
    int ks = tile >> 3, ct = tile & 7;
    int kd = ks * 32 + kbase + i;                 // K-dim index in [0,64)
    v = W[kd * KK + ct * 16 + col];               // B[k][n] = W[k][n]
  } else {
    int ks = tile >> 2, dt = tile & 3;
    int kd = ks * 32 + kbase + i;                 // K-dim index in [0,128)
    v = W[(dt * 16 + col) * KK + kd];             // B[k][d] = W[d][k]
  }
  (set ? B2 : B1)[m * 8192 + idx] = (bf)v;
}

// -------------------------------------------------- intent block via WMMA
// intl[n] = softmax(x[n] @ W) @ W^T.  One wave per 16-node tile, 8 waves/blk.
__global__ void __launch_bounds__(256)
k_intent(const float* __restrict__ x, const bf* __restrict__ B1,
         const bf* __restrict__ B2, float* __restrict__ intl, int n) {
  // per-wave probs staged in A-fragment layout: [ks<4][32 lanes][16] bf16
  __shared__ alignas(32) bf sPA[8][4 * 512];      // 8 * 4KB = 32KB

  const int wave  = threadIdx.x >> 5;
  const int lane  = threadIdx.x & 31;
  const int node0 = (blockIdx.x * 8 + wave) * 16;
  if (node0 >= n) return;

  // NUSERS % 16 == 0 so a tile never straddles the user/item boundary.
  const int m = (node0 < NUSERS) ? 0 : 1;
  const bf* __restrict__ W1 = B1 + m * 8192;
  const bf* __restrict__ W2 = B2 + m * 8192;

  const int row = lane & 15;        // A row (M) / output row base
  const int kb  = (lane >> 4) * 8;  // A-frag K base within 32-chunk
  const int col = lane & 15;
  const int hoff = (lane >> 4) << 3;  // +8 row offset for high half in C/D

  // ---- A fragments for GEMM1 from x, vectorized float4 loads (8 of them)
  const float4* __restrict__ xr = (const float4*)(x + (size_t)(node0 + row) * DIM);
  float4 t0 = xr[(kb >> 2) + 0],  t1 = xr[(kb >> 2) + 1];
  float4 t2 = xr[(kb >> 2) + 4],  t3 = xr[(kb >> 2) + 5];
  float4 t4 = xr[(kb >> 2) + 8],  t5 = xr[(kb >> 2) + 9];
  float4 t6 = xr[(kb >> 2) + 12], t7 = xr[(kb >> 2) + 13];
  v16bf a0, a1;
  a0[0]=(bf)t0.x; a0[1]=(bf)t0.y; a0[2]=(bf)t0.z; a0[3]=(bf)t0.w;
  a0[4]=(bf)t1.x; a0[5]=(bf)t1.y; a0[6]=(bf)t1.z; a0[7]=(bf)t1.w;
  a0[8]=(bf)t2.x; a0[9]=(bf)t2.y; a0[10]=(bf)t2.z; a0[11]=(bf)t2.w;
  a0[12]=(bf)t3.x; a0[13]=(bf)t3.y; a0[14]=(bf)t3.z; a0[15]=(bf)t3.w;
  a1[0]=(bf)t4.x; a1[1]=(bf)t4.y; a1[2]=(bf)t4.z; a1[3]=(bf)t4.w;
  a1[4]=(bf)t5.x; a1[5]=(bf)t5.y; a1[6]=(bf)t5.z; a1[7]=(bf)t5.w;
  a1[8]=(bf)t6.x; a1[9]=(bf)t6.y; a1[10]=(bf)t6.z; a1[11]=(bf)t6.w;
  a1[12]=(bf)t7.x; a1[13]=(bf)t7.y; a1[14]=(bf)t7.z; a1[15]=(bf)t7.w;

  // ---- GEMM1: logits(16x128) = X(16x64) @ W(64x128)
  v8f logit[8];
  #pragma unroll
  for (int ct = 0; ct < 8; ++ct) {
    v8f c = {};
    v16bf b0 = *(const v16bf*)(W1 + (0 * 8 + ct) * 512 + lane * 16);
    c = __builtin_amdgcn_wmma_f32_16x16x32_bf16(false, a0, false, b0, (short)0, c, false, false);
    v16bf b1 = *(const v16bf*)(W1 + (1 * 8 + ct) * 512 + lane * 16);
    c = __builtin_amdgcn_wmma_f32_16x16x32_bf16(false, a1, false, b1, (short)0, c, false, false);
    logit[ct] = c;
  }

  // ---- softmax per row over 128 cols; half-wave xor reductions.
  // Write probs into LDS directly in A-fragment layout for GEMM2.
  bf* __restrict__ PA = sPA[wave];
  const int h2 = (col >= 8) ? 1 : 0;   // target half for this lane's columns
  #pragma unroll
  for (int r = 0; r < 8; ++r) {
    float mx = -3.4e38f;
    #pragma unroll
    for (int ct = 0; ct < 8; ++ct) mx = fmaxf(mx, logit[ct][r]);
    #pragma unroll
    for (int off = 1; off < 16; off <<= 1) mx = fmaxf(mx, __shfl_xor(mx, off, 32));
    float ex[8], s = 0.f;
    #pragma unroll
    for (int ct = 0; ct < 8; ++ct) { ex[ct] = __expf(logit[ct][r] - mx); s += ex[ct]; }
    #pragma unroll
    for (int off = 1; off < 16; off <<= 1) s += __shfl_xor(s, off, 32);
    float inv = 1.f / s;
    const int prow = r + hoff;
    #pragma unroll
    for (int ct = 0; ct < 8; ++ct) {
      int k  = ct * 16 + col;
      int ks = k >> 5;
      int kk = k & 31;
      int i2 = (kk < 8) ? kk : (kk < 16) ? kk - 8 : (kk < 24) ? kk - 8 : kk - 16;
      PA[ks * 512 + (h2 * 16 + prow) * 16 + i2] = (bf)(ex[ct] * inv);
    }
  }
  // per-wave LDS staging (in-order per wave); drain DS before re-reading
  asm volatile("s_wait_dscnt 0" ::: "memory");

  // ---- GEMM2: out(16x64) = P(16x128) @ W^T(128x64)
  v16bf aF[4];
  #pragma unroll
  for (int ks = 0; ks < 4; ++ks)
    aF[ks] = *(const v16bf*)(PA + ks * 512 + lane * 16);

  #pragma unroll
  for (int dt = 0; dt < 4; ++dt) {
    v8f c = {};
    #pragma unroll
    for (int ks = 0; ks < 4; ++ks) {
      v16bf b = *(const v16bf*)(W2 + (ks * 4 + dt) * 512 + lane * 16);
      c = __builtin_amdgcn_wmma_f32_16x16x32_bf16(false, aF[ks], false, b, (short)0, c, false, false);
    }
    #pragma unroll
    for (int r = 0; r < 8; ++r) {
      int orow = node0 + r + hoff;
      intl[(size_t)orow * DIM + dt * 16 + col] = c[r];
    }
  }
}

// ------------------------------------------------------------------ driver
extern "C" void kernel_launch(void* const* d_in, const int* in_sizes, int n_in,
                              void* d_out, int out_size, void* d_ws, size_t ws_size,
                              hipStream_t stream) {
  const float* emb = (const float*)d_in[0];
  const float* uW  = (const float*)d_in[1];
  const float* iW  = (const float*)d_in[2];
  const int*   h   = (const int*)d_in[3];
  const int*   t   = (const int*)d_in[4];
  const int n = in_sizes[0] / DIM;     // 150000
  const int e = in_sizes[3];           // 1600000
  float* acc = (float*)d_out;

  // workspace carving, 256B-aligned slots (vector loads need >=32B alignment)
  char* w = (char*)d_ws;
  auto take = [&](size_t bytes) {
    char* p = w;
    w += (bytes + 255) & ~(size_t)255;
    return p;
  };
  const size_t ND = (size_t)n * DIM;
  float* xA   = (float*)take(ND * 4);
  float* xB   = (float*)take(ND * 4);
  float* gnn  = (float*)take(ND * 4);
  float* intl = (float*)take(ND * 4);
  float* gv   = (float*)take((size_t)e * 4);
  float* al   = (float*)take((size_t)e * 4);
  float* rs   = (float*)take((size_t)n * 4);
  float* invn = (float*)take((size_t)n * 4);
  bf*    B1   = (bf*)take(2 * 8192 * sizeof(bf));   // GEMM1 B frags
  bf*    B2   = (bf*)take(2 * 8192 * sizeof(bf));   // GEMM2 B frags

  const dim3 B(256);
  auto G = [](long cnt) { return dim3((unsigned)((cnt + 255) / 256)); };
  const dim3 gNodeW((unsigned)((n + 7) / 8));           // wave-per-node
  const dim3 gEdgeW((unsigned)((e + 7) / 8));           // wave-per-edge
  const dim3 gIntent((unsigned)(((n + 15) / 16 + 7) / 8));
  const long NDl = (long)ND;

  // acc = x = embeddings; swizzle intent matrices into fragment layout
  k_init<<<G(NDl), B, 0, stream>>>(emb, acc, xA, NDl);
  k_prepW<<<G(4 * 8192), B, 0, stream>>>(uW, iW, B1, B2);

  // symmetric-norm edge values
  k_zero<<<G(n), B, 0, stream>>>(rs, n);
  k_deg<<<G(e), B, 0, stream>>>(h, rs, e);
  k_dinvsqrt<<<G(n), B, 0, stream>>>(rs, invn, n);
  k_gvals<<<G(e), B, 0, stream>>>(invn, h, t, gv, e);

  float* x  = xA;
  float* xn = xB;
  for (int layer = 0; layer < 3; ++layer) {
    // gnn = G @ x
    k_zero<<<G(NDl), B, 0, stream>>>(gnn, NDl);
    k_spmm<<<G((long)e * DIM), B, 0, stream>>>(gv, h, t, x, gnn, e);

    // intl = softmax(x @ W) @ W^T   (WMMA)
    k_intent<<<gIntent, B, 0, stream>>>(x, B1, B2, intl, n);

    // xn accumulates gaa + iaa
    k_zero<<<G(NDl), B, 0, stream>>>(xn, NDl);

    // gaa: adaptive values from gnn
    k_invnorm<<<gNodeW, B, 0, stream>>>(gnn, invn, n);
    k_zero<<<G(n), B, 0, stream>>>(rs, n);
    k_alpha<<<gEdgeW, B, 0, stream>>>(gnn, invn, h, t, al, rs, e);
    k_scale<<<G(e), B, 0, stream>>>(al, rs, h, e);
    k_spmm<<<G((long)e * DIM), B, 0, stream>>>(al, h, t, x, xn, e);

    // iaa: adaptive values from intl
    k_invnorm<<<gNodeW, B, 0, stream>>>(intl, invn, n);
    k_zero<<<G(n), B, 0, stream>>>(rs, n);
    k_alpha<<<gEdgeW, B, 0, stream>>>(intl, invn, h, t, al, rs, e);
    k_scale<<<G(e), B, 0, stream>>>(al, rs, h, e);
    k_spmm<<<G((long)e * DIM), B, 0, stream>>>(al, h, t, x, xn, e);

    // xn = xn + gnn + intl + x ; acc += xn ; swap
    k_combine<<<G(NDl), B, 0, stream>>>(xn, gnn, intl, x, acc, NDl);
    float* tmp = x; x = xn; xn = tmp;
  }
}
